// mLSTM_43301860278402
// MI455X (gfx1250) — matrix operand order repaired and checked by reading, not compiled, not run on hardware
//
#include <hip/hip_runtime.h>
#include <math.h>

typedef __attribute__((ext_vector_type(16))) _Float16 v16h;
typedef __attribute__((ext_vector_type(8)))  float    v8f;
typedef __attribute__((ext_vector_type(4)))  int      v4i;

#define HID     256
#define THREE_H 768
#define NCOLS   1536          // [gates_x(768) | q(256) | k(256) | v(256)]
#define TLEN    512
#define NBATCH  8
#define BT      (NBATCH * TLEN)   // 4096 rows
#define CT_STRIDE 260             // 256 + 4 pad -> bank rotation 4 mod 64

#if defined(__has_builtin)
#if __has_builtin(__builtin_amdgcn_global_load_async_to_lds_b128)
#define HAVE_ASYNC_LDS 1
#endif
#endif
#ifndef HAVE_ASYNC_LDS
#define HAVE_ASYNC_LDS 0
#endif

// ---------------------------------------------------------------------------
// Phase A: P[m, 0:1536] = concat(x@w_ih.T + b, x@wq.T+bq, x@wk.T+bk, x@wv.T+bv)
// WMMA f32<-f16. Each wave: 1 M-tile x 4 N-tiles (A fragment reused 4x).
// ---------------------------------------------------------------------------

__device__ __forceinline__ const float* wrow(int g, const float* w_ih, const float* wq,
                                             const float* wk, const float* wv) {
  if (g < THREE_H)            return w_ih + (size_t)g * HID;
  if (g < THREE_H + HID)      return wq + (size_t)(g - THREE_H) * HID;
  if (g < THREE_H + 2 * HID)  return wk + (size_t)(g - THREE_H - HID) * HID;
  return wv + (size_t)(g - THREE_H - 2 * HID) * HID;
}

__device__ __forceinline__ float bval(int g, const float* b, const float* bq,
                                      const float* bk, const float* bv) {
  if (g < THREE_H)            return b[g];
  if (g < THREE_H + HID)      return bq[g - THREE_H];
  if (g < THREE_H + 2 * HID)  return bk[g - THREE_H - HID];
  return bv[g - THREE_H - 2 * HID];
}

__device__ __forceinline__ v16h pack16(float4 a, float4 b, float4 c, float4 d) {
  v16h r;
  r[0]=(_Float16)a.x; r[1]=(_Float16)a.y; r[2]=(_Float16)a.z; r[3]=(_Float16)a.w;
  r[4]=(_Float16)b.x; r[5]=(_Float16)b.y; r[6]=(_Float16)b.z; r[7]=(_Float16)b.w;
  r[8]=(_Float16)c.x; r[9]=(_Float16)c.y; r[10]=(_Float16)c.z; r[11]=(_Float16)c.w;
  r[12]=(_Float16)d.x; r[13]=(_Float16)d.y; r[14]=(_Float16)d.z; r[15]=(_Float16)d.w;
  return r;
}

__global__ __launch_bounds__(256) void proj_gemm(
    const float* __restrict__ X,
    const float* __restrict__ w_ih, const float* __restrict__ b,
    const float* __restrict__ wq,   const float* __restrict__ bq,
    const float* __restrict__ wk,   const float* __restrict__ bk,
    const float* __restrict__ wv,   const float* __restrict__ bv,
    float* __restrict__ P) {
  const int lane = threadIdx.x & 31;
  const int wave = threadIdx.x >> 5;
  const int m0 = blockIdx.x * 16;              // 256 row tiles
  const int n0 = blockIdx.y * 512 + wave * 64; // 3 blocks x 8 waves x 64 cols

  const int l15 = lane & 15;
  const int hi  = lane >> 4;  // 0: lanes 0-15, 1: lanes 16-31
  const float* xr = X + (size_t)(m0 + l15) * HID;

  int gcol[4];
  const float* wr[4];
#pragma unroll
  for (int nt = 0; nt < 4; ++nt) {
    gcol[nt] = n0 + nt * 16 + l15;
    wr[nt]   = wrow(gcol[nt], w_ih, wq, wk, wv);
  }

  const v8f vzero = {};
  v8f acc[4] = {vzero, vzero, vzero, vzero};

  for (int k0 = 0; k0 < HID; k0 += 32) {
    // A (16x32 f16): lanes 0-15 hold K = k0+{0..7,16..23}; lanes 16-31: +8
    const int ka = k0 + hi * 8;
    v16h a = pack16(*(const float4*)(xr + ka),      *(const float4*)(xr + ka + 4),
                    *(const float4*)(xr + ka + 16), *(const float4*)(xr + ka + 20));
    // B (32x16 f16): lanes 0-15 hold K = k0+{0..7,16..23} of their column; lanes 16-31: +8
    const int kb = k0 + hi * 8;
#pragma unroll
    for (int nt = 0; nt < 4; ++nt) {
      v16h bfr = pack16(*(const float4*)(wr[nt] + kb),     *(const float4*)(wr[nt] + kb + 4),
                        *(const float4*)(wr[nt] + kb + 16), *(const float4*)(wr[nt] + kb + 20));
      acc[nt] = __builtin_amdgcn_wmma_f32_16x16x32_f16(false, a, false, bfr,
                                                       (short)0, acc[nt], false, false);
    }
  }
  const int orow0 = m0 + hi * 8;  // C/D layout: vgpr r -> M = r (+8 for hi lanes)
#pragma unroll
  for (int nt = 0; nt < 4; ++nt) {
    const float bias = bval(gcol[nt], b, bq, bk, bv);
#pragma unroll
    for (int r = 0; r < 8; ++r)
      P[(size_t)(orow0 + r) * NCOLS + gcol[nt]] = acc[nt][r] + bias;
  }
}

// ---------------------------------------------------------------------------
// Phase B: sequential scan. One workgroup per batch element; C^T lives in LDS.
// P-row double-buffered in LDS via async global->LDS prefetch.
// ---------------------------------------------------------------------------

#if HAVE_ASYNC_LDS
typedef __attribute__((address_space(1))) v4i* async_gptr;
typedef __attribute__((address_space(3))) v4i* async_lptr;
#endif

__device__ __forceinline__ void prefetch_row(const float* gsrc, float* ldst, int tid) {
  if (tid < 384) {  // 384 threads x 16B = 6144B = one P row
#if HAVE_ASYNC_LDS
    __builtin_amdgcn_global_load_async_to_lds_b128(
        (async_gptr)(gsrc + tid * 4), (async_lptr)(ldst + tid * 4), 0, 0);
#else
    *(float4*)(ldst + tid * 4) = *(const float4*)(gsrc + tid * 4);
#endif
  }
}

__device__ __forceinline__ void wait_async_done() {
#if HAVE_ASYNC_LDS
#if __has_builtin(__builtin_amdgcn_s_wait_asynccnt)
  __builtin_amdgcn_s_wait_asynccnt(0);
#else
  asm volatile("s_wait_asynccnt 0x0" ::: "memory");
#endif
#endif
}

__global__ __launch_bounds__(1024) void mlstm_scan(
    const float* __restrict__ P, const float* __restrict__ w_hh,
    float* __restrict__ Y) {
  extern __shared__ float smem[];
  float* CT   = smem;                    // [256][260] : CT[j][i] = C[i][j]
  float* f_a  = CT + 256 * CT_STRIDE;    // [256] exp(f-gate)
  float* a_a  = f_a + 256;               // [256] exp(i-gate) * v
  float* o_a  = a_a + 256;               // [256] sigmoid(o-gate)
  float* h_a  = o_a + 256;               // [256] h
  float* part = h_a + 256;               // [4][256] reduction partials
  float* pbuf = part + 1024;             // [2][1536] P-row double buffer

  const int tid   = threadIdx.x;
  const int batch = blockIdx.x;
  const int lane  = tid & 31;
  const int wave  = tid >> 5;

  const float* Pbase = P + (size_t)batch * TLEN * NCOLS;
  prefetch_row(Pbase, pbuf, tid);  // row 0 while we zero C

  for (int idx = tid; idx < 256 * CT_STRIDE; idx += 1024) CT[idx] = 0.f;
  if (tid < 256) h_a[tid] = 0.f;
  wait_async_done();
  __syncthreads();

  const int j  = tid & 255;   // owned column of C
  const int rg = tid >> 8;    // row group 0..3 (64 rows each)
  const int i0 = rg * 64;
  float* ctrow = CT + (size_t)j * CT_STRIDE;
  const int gbase = wave * 24;  // 32 waves x 24 = 768 gate outputs

  for (int tt = 0; tt < TLEN; ++tt) {
    const float* Prow = pbuf + (tt & 1) * NCOLS;  // LDS-resident row
    if (tt + 1 < TLEN)
      prefetch_row(Pbase + (size_t)(tt + 1) * NCOLS, pbuf + ((tt + 1) & 1) * NCOLS, tid);

    // h into registers (distributed across the wave)
    float hreg[8];
#pragma unroll
    for (int u = 0; u < 8; ++u) hreg[u] = h_a[u * 32 + lane];

    // gh[g] = h @ w_hh[g,:]  — wave-cooperative dot + butterfly reduce
    float mygh = 0.f;
    for (int idx = 0; idx < 24; ++idx) {
      const float* wrp = w_hh + (size_t)(gbase + idx) * HID;
      float s = 0.f;
#pragma unroll
      for (int u = 0; u < 8; ++u) s += wrp[u * 32 + lane] * hreg[u];
#pragma unroll
      for (int m = 16; m >= 1; m >>= 1) s += __shfl_xor(s, m, 32);
      if (lane == idx) mygh = s;
    }
    // gate nonlinearities: lanes 0..23 handle g = gbase+lane in parallel
    if (lane < 24) {
      const int g = gbase + lane;
      const float pv = Prow[g] + mygh;
      if (g < 256) {                           // i-gate (row index)
        a_a[g] = expf(pv) * Prow[1280 + g];    // exp(i) * v
      } else if (g < 512) {                    // f-gate
        f_a[g - 256] = expf(pv);
      } else {                                 // o-gate (col index)
        o_a[g - 512] = 1.f / (1.f + expf(-pv));
      }
    }
    __syncthreads();

    // C[i,j] = f[i]*C[i,j] + a[i]*k[j];  partial h_j = sum_i q[i]*C[i,j]
    const float kj = Prow[1024 + j];
    const float* qrow = Prow + 768;
    float partial = 0.f;
#pragma unroll
    for (int cc = 0; cc < 16; ++cc) {
      const int i = i0 + cc * 4;
      float4 c4 = *(float4*)(ctrow + i);                 // ds_load_b128
      float4 f4 = *(const float4*)(f_a + i);             // broadcast
      float4 a4 = *(const float4*)(a_a + i);
      float4 q4 = *(const float4*)(qrow + i);
      c4.x = f4.x * c4.x + a4.x * kj;
      c4.y = f4.y * c4.y + a4.y * kj;
      c4.z = f4.z * c4.z + a4.z * kj;
      c4.w = f4.w * c4.w + a4.w * kj;
      *(float4*)(ctrow + i) = c4;                        // ds_store_b128
      partial += q4.x * c4.x + q4.y * c4.y + q4.z * c4.z + q4.w * c4.w;
    }
    part[rg * 256 + j] = partial;
    __syncthreads();

    if (tid < 256) {
      const float hv = o_a[tid] *
          (part[tid] + part[256 + tid] + part[512 + tid] + part[768 + tid]);
      h_a[tid] = hv;
      Y[((size_t)batch * TLEN + tt) * HID + tid] = hv;
    }
    wait_async_done();  // next row resident before anyone reads it
    __syncthreads();
  }
}

// ---------------------------------------------------------------------------

extern "C" void kernel_launch(void* const* d_in, const int* in_sizes, int n_in,
                              void* d_out, int out_size, void* d_ws, size_t ws_size,
                              hipStream_t stream) {
  (void)in_sizes; (void)n_in; (void)out_size; (void)ws_size;
  const float* x     = (const float*)d_in[0];
  const float* w_ih0 = (const float*)d_in[1];
  const float* w_hh0 = (const float*)d_in[2];
  const float* b0    = (const float*)d_in[3];
  const float* wq0   = (const float*)d_in[4];
  const float* bq0   = (const float*)d_in[5];
  const float* wk0   = (const float*)d_in[6];
  const float* bk0   = (const float*)d_in[7];
  const float* wv0   = (const float*)d_in[8];
  const float* bv0   = (const float*)d_in[9];
  const float* w_ih1 = (const float*)d_in[10];
  const float* w_hh1 = (const float*)d_in[11];
  const float* b1    = (const float*)d_in[12];
  const float* wq1   = (const float*)d_in[13];
  const float* bq1   = (const float*)d_in[14];
  const float* wk1   = (const float*)d_in[15];
  const float* bk1   = (const float*)d_in[16];
  const float* wv1   = (const float*)d_in[17];
  const float* bv1   = (const float*)d_in[18];

  float* P   = (float*)d_ws;                                  // 4096*1536*4 = 25165824 B
  float* X1  = (float*)((char*)d_ws + (size_t)BT * NCOLS * 4);// 4096*256*4  =  4194304 B
  float* out = (float*)d_out;

  const size_t lds_bytes =
      (256 * CT_STRIDE + 4 * 256 + 4 * 256 + 2 * NCOLS) * sizeof(float);  // 286720
  (void)hipFuncSetAttribute(reinterpret_cast<const void*>(mlstm_scan),
                            hipFuncAttributeMaxDynamicSharedMemorySize,
                            (int)lds_bytes);

  dim3 ggrid(BT / 16, NCOLS / 512), gblk(256);

  // Layer 0
  proj_gemm<<<ggrid, gblk, 0, stream>>>(x, w_ih0, b0, wq0, bq0, wk0, bk0, wv0, bv0, P);
  mlstm_scan<<<NBATCH, 1024, lds_bytes, stream>>>(P, w_hh0, X1);
  // Layer 1
  proj_gemm<<<ggrid, gblk, 0, stream>>>(X1, w_ih1, b1, wq1, bq1, wk1, bk1, wv1, bv1, P);
  mlstm_scan<<<NBATCH, 1024, lds_bytes, stream>>>(P, w_hh1, out);
}
